// PillarContext3D_def_53498112639177
// MI455X (gfx1250) — matrix operand order repaired and checked by reading, not compiled, not run on hardware
//
#include <hip/hip_runtime.h>
#include <cstdint>
#include <cstddef>

// ---------------- problem constants ----------------
#define Bb   2
#define Pp   12000
#define Cc   64
#define Kk   1024
#define NSs  16
#define NXx  432
#define NYy  496

#define XOFF   0.08f
#define YOFF  (-39.60f)
#define ZOFF  (-1.0f)
#define VXc    0.16f
#define VYc    0.16f
#define VZc    4.0f
#define RAD2   2.56f     /* 1.6^2  */
#define DRAD2 10.24f     /* 3.2^2  */
#define DRAD   3.2f
#define EPSf   1e-5f
#define NEGf  (-1e9f)
#define SCALEf 0.125f    /* 1/sqrt(64) */

typedef __bf16 bf16;
typedef __attribute__((ext_vector_type(16))) __bf16 v16bf;
typedef __attribute__((ext_vector_type(8)))  float  v8f;

// ---- CDNA5 async global->LDS copy (guarded; falls back to b128 copies) ----
// Probe-verified: params are GCC-vector int4 pointers in AS1 / AS3.
typedef int a4i __attribute__((vector_size(16)));
#if defined(__HIP_DEVICE_COMPILE__) &&                                        \
    __has_builtin(__builtin_amdgcn_global_load_async_to_lds_b128) &&          \
    __has_builtin(__builtin_amdgcn_s_wait_asynccnt)
#define USE_ASYNC 1
__device__ __forceinline__ void async_copy16(const void* g, void* l) {
  __builtin_amdgcn_global_load_async_to_lds_b128(
      (__attribute__((address_space(1))) a4i*)(void*)g,
      (__attribute__((address_space(3))) a4i*)l, 0, 0);
}
__device__ __forceinline__ void async_wait0() {
  __builtin_amdgcn_s_wait_asynccnt(0);
}
#else
#define USE_ASYNC 0
__device__ __forceinline__ void async_copy16(const void* g, void* l) {
  *(uint4*)l = *(const uint4*)g;
}
__device__ __forceinline__ void async_wait0() {}
#endif

// ---- CDNA5 WMMA bf16 fragment helpers (ISA 7.12.2 layouts) ----
// A 16x32 (MxK): lane holds row M=lane&15; elems = two 8-elem runs at
//   K = ((lane&16)?8:0) + {0..7} and +16..23  -> two 16B loads.
// B 32x16 (KxN): lane holds col N=lane&15; elems = two 8-elem runs at
//   K = ((lane&16)?8:0) + {0..7} and +16..23  -> two 16B loads.
// C/D f32 16x16: lane<16 -> N=lane, M=r ; lane>=16 -> N=lane-16, M=8+r.
union Frag {
  v16bf v;
  uint4 u[2];
};
union U8x {
  uint4 q;
  bf16  h[8];
};

__device__ __forceinline__ v16bf load_frag2(const bf16* p0, const bf16* p1) {
  Frag f;
  f.u[0] = *(const uint4*)p0;
  f.u[1] = *(const uint4*)p1;
  return f.v;
}

__device__ __forceinline__ v8f wmma_bf16(v16bf a, v16bf b, v8f c) {
  return __builtin_amdgcn_wmma_f32_16x16x32_bf16(false, a, false, b, (short)0, c,
                                                 false, false);
}

// =====================================================================
// Stage 1: pillar centers from voxel coords
// =====================================================================
__global__ void centers_kernel(const int* __restrict__ coords,
                               float* __restrict__ centers) {
  int t = blockIdx.x * blockDim.x + threadIdx.x;
  if (t >= Bb * Pp) return;
  const int* c = coords + (size_t)t * 4;
  centers[t * 3 + 0] = (float)c[3] * VXc + XOFF;
  centers[t * 3 + 1] = (float)c[2] * VYc + YOFF;
  centers[t * 3 + 2] = (float)c[1] * VZc + ZOFF;
}

// =====================================================================
// Stage 2: furthest-point sampling (one WG per batch, dists in LDS)
// =====================================================================
__global__ void fps_kernel(const float* __restrict__ centers,
                           int* __restrict__ kpidx, float* __restrict__ kp) {
  const int b = blockIdx.x, tid = threadIdx.x;  // 1024 threads
  const float* xyz = centers + (size_t)b * Pp * 3;
  __shared__ float dists[Pp];          // 48 KB of 320 KB WGP LDS
  __shared__ float rd[1024];
  __shared__ int   ri[1024];
  __shared__ float curx[3];
  __shared__ int   cur;
  for (int p = tid; p < Pp; p += 1024) dists[p] = 1e10f;
  if (tid == 0) {
    cur = 0;
    kpidx[b * Kk] = 0;
    kp[(size_t)b * Kk * 3 + 0] = xyz[0];
    kp[(size_t)b * Kk * 3 + 1] = xyz[1];
    kp[(size_t)b * Kk * 3 + 2] = xyz[2];
  }
  __syncthreads();
  for (int k = 1; k < Kk; ++k) {
    if (tid < 3) curx[tid] = xyz[cur * 3 + tid];
    __syncthreads();
    float best = -1.0f; int bi = 0;
    for (int p = tid; p < Pp; p += 1024) {
      float dx = xyz[p * 3 + 0] - curx[0];
      float dy = xyz[p * 3 + 1] - curx[1];
      float dz = xyz[p * 3 + 2] - curx[2];
      float d = dx * dx + dy * dy + dz * dz;
      float nd = fminf(dists[p], d);
      dists[p] = nd;
      if (nd > best) { best = nd; bi = p; }
    }
    rd[tid] = best; ri[tid] = bi;
    __syncthreads();
    for (int s = 512; s > 0; s >>= 1) {
      if (tid < s) {
        if (rd[tid + s] > rd[tid] ||
            (rd[tid + s] == rd[tid] && ri[tid + s] < ri[tid])) {
          rd[tid] = rd[tid + s]; ri[tid] = ri[tid + s];
        }
      }
      __syncthreads();
    }
    if (tid == 0) {
      cur = ri[0];
      kpidx[b * Kk + k] = cur;
      kp[((size_t)b * Kk + k) * 3 + 0] = xyz[cur * 3 + 0];
      kp[((size_t)b * Kk + k) * 3 + 1] = xyz[cur * 3 + 1];
      kp[((size_t)b * Kk + k) * 3 + 2] = xyz[cur * 3 + 2];
    }
    __syncthreads();
  }
}

// =====================================================================
// Stage 3/5: exact nearest-16 of 12000 (matches top_k(-d2,16)) + radius mask
// =====================================================================
__global__ void group_topk(const float* __restrict__ xyz,
                           const float* __restrict__ query,
                           int* __restrict__ idx, int* __restrict__ valid,
                           float r2) {
  const int b = blockIdx.z, k = blockIdx.x, tid = threadIdx.x;  // 256 thr
  const float* pts = xyz + (size_t)b * Pp * 3;
  const size_t qrow = (size_t)b * Kk + k;
  const float qx = query[qrow * 3 + 0];
  const float qy = query[qrow * 3 + 1];
  const float qz = query[qrow * 3 + 2];
  __shared__ int   ch_i[NSs];
  __shared__ float ch_d[NSs];
  __shared__ float rd[256];
  __shared__ int   ri[256];
  for (int r = 0; r < NSs; ++r) {
    float best = 3.0e38f; int bi = Pp;
    for (int p = tid; p < Pp; p += 256) {
      bool used = false;
      for (int j = 0; j < r; ++j) used |= (ch_i[j] == p);
      if (used) continue;
      float dx = pts[p * 3 + 0] - qx;
      float dy = pts[p * 3 + 1] - qy;
      float dz = pts[p * 3 + 2] - qz;
      float d = dx * dx + dy * dy + dz * dz;
      if (d < best || (d == best && p < bi)) { best = d; bi = p; }
    }
    rd[tid] = best; ri[tid] = bi;
    __syncthreads();
    for (int s = 128; s > 0; s >>= 1) {
      if (tid < s) {
        float od = rd[tid + s]; int oi = ri[tid + s];
        if (od < rd[tid] || (od == rd[tid] && oi < ri[tid])) {
          rd[tid] = od; ri[tid] = oi;
        }
      }
      __syncthreads();
    }
    if (tid == 0) { ch_i[r] = ri[0]; ch_d[r] = rd[0]; }
    __syncthreads();
  }
  if (tid < NSs) {
    idx[qrow * NSs + tid]   = ch_i[tid];
    valid[qrow * NSs + tid] = (ch_d[tid] <= r2) ? 1 : 0;
  }
}

// =====================================================================
// Stage 4: deformation offsets (g0 @ w_off, masked mean, tanh, clip)
// =====================================================================
__global__ void deform_kernel(const float* __restrict__ centers,
                              const float* __restrict__ feats,
                              const float* __restrict__ kp,
                              const int* __restrict__ idx,
                              const int* __restrict__ valid,
                              const float* __restrict__ w_off,
                              float* __restrict__ defxyz) {
  int t = blockIdx.x * blockDim.x + threadIdx.x;
  if (t >= Bb * Kk) return;
  int b = t / Kk;
  float cx = kp[t * 3 + 0], cy = kp[t * 3 + 1], cz = kp[t * 3 + 2];
  float acc0 = 0.f, acc1 = 0.f, acc2 = 0.f;
  int cnt = 0;
  for (int ns = 0; ns < NSs; ++ns) {
    if (!valid[t * NSs + ns]) continue;
    int p = idx[t * NSs + ns];
    const float* pt = centers + ((size_t)b * Pp + p) * 3;
    float r0 = pt[0] - cx, r1 = pt[1] - cy, r2v = pt[2] - cz;
    float s0 = r0 * w_off[0] + r1 * w_off[3] + r2v * w_off[6];
    float s1 = r0 * w_off[1] + r1 * w_off[4] + r2v * w_off[7];
    float s2 = r0 * w_off[2] + r1 * w_off[5] + r2v * w_off[8];
    const float* f = feats + ((size_t)b * Pp + p) * Cc;
    for (int j = 0; j < Cc; ++j) {
      float fv = f[j];
      const float* wr = w_off + (size_t)(3 + j) * 3;
      s0 += fv * wr[0]; s1 += fv * wr[1]; s2 += fv * wr[2];
    }
    acc0 += s0; acc1 += s1; acc2 += s2; ++cnt;
  }
  float den = (float)(cnt > 0 ? cnt : 1);
  float ox = tanhf(acc0 / den) * DRAD;
  float oy = tanhf(acc1 / den) * DRAD;
  float oz = tanhf(acc2 / den) * DRAD;
  defxyz[t * 3 + 0] = fminf(fmaxf(cx + ox, 0.0f),    69.12f);
  defxyz[t * 3 + 1] = fminf(fmaxf(cy + oy, -39.68f), 39.68f);
  defxyz[t * 3 + 2] = fminf(fmaxf(cz + oz, -3.0f),   1.0f);
}

// =====================================================================
// Stage 6: shared MLP + masked max-pool (one WG of 64 thr per keypoint)
// =====================================================================
__global__ void mlp_kernel(const float* __restrict__ centers,
                           const float* __restrict__ feats,
                           const float* __restrict__ defxyz,
                           const int* __restrict__ idx,
                           const int* __restrict__ valid,
                           const float* __restrict__ w1,
                           const float* __restrict__ b1,
                           const float* __restrict__ w2,
                           const float* __restrict__ b2,
                           float* __restrict__ local) {
  const int b = blockIdx.z, k = blockIdx.x, c = threadIdx.x;  // 64 thr
  const size_t t = (size_t)b * Kk + k;
  __shared__ float nb[NSs][Cc + 3];
  __shared__ float h1[NSs][Cc];
  const float qx = defxyz[t * 3 + 0];
  const float qy = defxyz[t * 3 + 1];
  const float qz = defxyz[t * 3 + 2];
  for (int i = c; i < NSs * (Cc + 3); i += Cc) {
    int ns = i / (Cc + 3), j = i % (Cc + 3);
    int p = idx[t * NSs + ns];
    float v;
    if (j == 0)      v = centers[((size_t)b * Pp + p) * 3 + 0] - qx;
    else if (j == 1) v = centers[((size_t)b * Pp + p) * 3 + 1] - qy;
    else if (j == 2) v = centers[((size_t)b * Pp + p) * 3 + 2] - qz;
    else             v = feats[((size_t)b * Pp + p) * Cc + (j - 3)];
    nb[ns][j] = v;
  }
  __syncthreads();
  for (int ns = 0; ns < NSs; ++ns) {
    float s = b1[c];
    for (int j = 0; j < Cc + 3; ++j) s += nb[ns][j] * w1[j * Cc + c];
    h1[ns][c] = fmaxf(s, 0.f);
  }
  __syncthreads();
  float pooled = NEGf;
  for (int ns = 0; ns < NSs; ++ns) {
    float s = b2[c];
    for (int j = 0; j < Cc; ++j) s += h1[ns][j] * w2[j * Cc + c];
    s = fmaxf(s, 0.f);
    if (valid[t * NSs + ns]) pooled = fmaxf(pooled, s);
  }
  local[t * Cc + c] = (pooled > NEGf * 0.5f) ? pooled : 0.f;
}

// =====================================================================
// fp32 -> bf16 conversion
// =====================================================================
__global__ void f32_to_bf16(const float* __restrict__ x, bf16* __restrict__ y,
                            int n) {
  int i = blockIdx.x * blockDim.x + threadIdx.x;
  if (i < n) y[i] = (bf16)x[i];
}

// =====================================================================
// WMMA GEMM: [M,64] @ [64,64]; W staged transposed in LDS so B-fragments
// are 2x ds_load_b128; A-fragments are 2x global b128. Output mode is a
// compile-time template (branch-free epilogue).
// =====================================================================
template <bool WF32, bool WBF16>
__global__ void gemm64(const bf16* __restrict__ A, const bf16* __restrict__ W,
                       float* __restrict__ Of, bf16* __restrict__ Ob, int M) {
  const int bz = blockIdx.z;
  A += (size_t)bz * M * Cc;
  if (WF32)  Of += (size_t)bz * M * Cc;
  if (WBF16) Ob += (size_t)bz * M * Cc;
  const int tid = threadIdx.x;                  // 128 threads
  const int wave = tid >> 5, lane = tid & 31;
  __shared__ __align__(16) bf16 Wt[64 * 64];    // Wt[n*64+k]
  for (int i = tid * 8; i < 64 * 64; i += 128 * 8) {
    int kdim = i >> 6, n0 = i & 63;
    U8x u; u.q = *(const uint4*)(W + i);
#pragma unroll
    for (int j = 0; j < 8; ++j) Wt[(n0 + j) * 64 + kdim] = u.h[j];
  }
  __syncthreads();
  const int tiles = M >> 4;
  int tile = blockIdx.x * 4 + wave;
  if (tile >= tiles) tile = tiles - 1;  // duplicate work keeps result identical
  const int kh_a = (lane & 16) ? 8 : 0;
  const int kh_b = (lane & 16) ? 8 : 0;
  v16bf a[2];
#pragma unroll
  for (int ch = 0; ch < 2; ++ch) {
    const bf16* p = A + (size_t)(tile * 16 + (lane & 15)) * Cc + ch * 32 + kh_a;
    a[ch] = load_frag2(p, p + 16);
  }
#pragma unroll
  for (int nt = 0; nt < 4; ++nt) {
    v8f acc = {0.f, 0.f, 0.f, 0.f, 0.f, 0.f, 0.f, 0.f};
#pragma unroll
    for (int ch = 0; ch < 2; ++ch) {
      const bf16* p = &Wt[(nt * 16 + (lane & 15)) * 64 + ch * 32 + kh_b];
      acc = wmma_bf16(a[ch], load_frag2(p, p + 16), acc);
    }
#pragma unroll
    for (int r = 0; r < 8; ++r) {
      size_t o = (size_t)(tile * 16 + r + ((lane & 16) ? 8 : 0)) * Cc +
                 nt * 16 + (lane & 15);
      if (WF32)  Of[o] = acc[r];
      if (WBF16) Ob[o] = (bf16)acc[r];
    }
  }
}

// =====================================================================
// Flash-attention: O = softmax(Q Kt^T / 8) V, bf16 in / bf16 out.
// 8 waves/block share double-buffered K tile (async global->LDS copy,
// prefetching tile n+1 during compute) and a transposed V tile in LDS.
// Scores live entirely in per-wave LDS slices; 8 WMMAs per 32-key step.
// =====================================================================
#define FAW 8
__global__ void flash_attn(const bf16* __restrict__ Q,
                           const bf16* __restrict__ Kt,
                           const bf16* __restrict__ V,
                           bf16* __restrict__ O, int M, int N) {
  const int bz = blockIdx.z;
  Q  += (size_t)bz * M * Cc;
  O  += (size_t)bz * M * Cc;
  Kt += (size_t)bz * N * Cc;
  V  += (size_t)bz * N * Cc;
  const int tid = threadIdx.x;                  // 256 threads
  const int wave = tid >> 5, lane = tid & 31;
  const int ntq = M >> 4;
  int tile = blockIdx.x * FAW + wave;
  if (tile >= ntq) tile = ntq - 1;  // duplicate work: keeps barriers uniform

  __shared__ __align__(16) bf16  sK[2][32 * 64];   // row-major K tile
  __shared__ __align__(16) bf16  sVt[2][64 * 32];  // transposed V tile
  __shared__ __align__(16) float sS[FAW][16][32];
  __shared__ __align__(16) bf16  sP[FAW][16][32];
  __shared__ float sStat[FAW][16][2];  // [0]=corr, [1]=l

  const int kh_a = (lane & 16) ? 8 : 0;
  const int kh_b = (lane & 16) ? 8 : 0;

  auto stageKV = [&](int keybase, int buf) {
    // K tile: 4 KB = 256 threads x 16 B (async on CDNA5, b128 fallback)
    async_copy16((const char*)(Kt + (size_t)keybase * Cc) + tid * 16,
                 (char*)&sK[buf][0] + tid * 16);
    // V tile transposed: coalesced 16B read, 8 ds_store_b16 scatter
    int key = tid >> 3;
    int c0 = (tid & 7) * 8;
    U8x u; u.q = *(const uint4*)(V + (size_t)(keybase + key) * Cc + c0);
#pragma unroll
    for (int j = 0; j < 8; ++j) sVt[buf][(c0 + j) * 32 + key] = u.h[j];
  };

  // q fragments (reused across all key tiles): 2x b128 each
  v16bf qa[2];
#pragma unroll
  for (int ch = 0; ch < 2; ++ch) {
    const bf16* p = Q + (size_t)(tile * 16 + (lane & 15)) * Cc + ch * 32 + kh_a;
    qa[ch] = load_frag2(p, p + 16);
  }
  v8f acc[4];
#pragma unroll
  for (int t = 0; t < 4; ++t)
#pragma unroll
    for (int r = 0; r < 8; ++r) acc[t][r] = 0.f;

  float m_i = -3.0e38f, l_i = 0.f;  // lanes 0..15 own row = lane

  stageKV(0, 0);
  async_wait0();
  __syncthreads();

  const int nt = N >> 5;
  for (int it = 0; it < nt; ++it) {
    const int cur = it & 1;
    if (it + 1 < nt) stageKV((it + 1) * 32, cur ^ 1);  // prefetch next tile
    // ---- S = Q @ K^T (two 16x16 tiles, K-dim 64 = 2 chunks)
#pragma unroll
    for (int t = 0; t < 2; ++t) {
      v8f s = {0.f, 0.f, 0.f, 0.f, 0.f, 0.f, 0.f, 0.f};
#pragma unroll
      for (int ch = 0; ch < 2; ++ch) {
        const bf16* p =
            &sK[cur][(t * 16 + (lane & 15)) * 64 + ch * 32 + kh_b];
        s = wmma_bf16(qa[ch], load_frag2(p, p + 16), s);
      }
#pragma unroll
      for (int r = 0; r < 8; ++r)
        sS[wave][r + ((lane & 16) ? 8 : 0)][t * 16 + (lane & 15)] = s[r];
    }
    __syncthreads();
    // ---- online softmax (lanes 0..15 handle one row each)
    if (lane < 16) {
      const int row = lane;
      float mx = m_i;
      for (int j = 0; j < 32; ++j) mx = fmaxf(mx, sS[wave][row][j] * SCALEf);
      float corr = __expf(m_i - mx);
      float sum = l_i * corr;
      for (int j = 0; j < 32; ++j) {
        float p = __expf(sS[wave][row][j] * SCALEf - mx);
        sP[wave][row][j] = (bf16)p;
        sum += p;
      }
      m_i = mx; l_i = sum;
      sStat[wave][row][0] = corr;
    }
    __syncthreads();
    // ---- rescale accumulator + O += P @ V
    const bf16* pp = &sP[wave][lane & 15][kh_a];
    v16bf pa = load_frag2(pp, pp + 16);
#pragma unroll
    for (int t = 0; t < 4; ++t) {
#pragma unroll
      for (int r = 0; r < 8; ++r)
        acc[t][r] *= sStat[wave][r + ((lane & 16) ? 8 : 0)][0];
      const bf16* p = &sVt[cur][(t * 16 + (lane & 15)) * 32 + kh_b];
      acc[t] = wmma_bf16(pa, load_frag2(p, p + 16), acc[t]);
    }
    if (it + 1 < nt) async_wait0();
    __syncthreads();  // all reads of buffer `cur` done; next tile resident
  }
  if (lane < 16) sStat[wave][lane][1] = l_i;
  __syncthreads();
#pragma unroll
  for (int t = 0; t < 4; ++t)
#pragma unroll
    for (int r = 0; r < 8; ++r) {
      int rl = r + ((lane & 16) ? 8 : 0);
      float l = sStat[wave][rl][1];
      O[(size_t)(tile * 16 + rl) * Cc + t * 16 + (lane & 15)] =
          (bf16)(acc[t][r] / l);
    }
}

// =====================================================================
// GroupNorm(num_groups=1) over whole [M,C] sample + affine + residual
// =====================================================================
__global__ void gn_reduce(const float* __restrict__ x, float* __restrict__ stats,
                          int M) {
  const int b = blockIdx.x, tid = threadIdx.x;
  const float* p = x + (size_t)b * M * Cc;
  __shared__ float s1[256], s2[256];
  float a = 0.f, q = 0.f;
  for (int i = tid; i < M * Cc; i += 256) {
    float v = p[i];
    a += v; q += v * v;
  }
  s1[tid] = a; s2[tid] = q;
  __syncthreads();
  for (int s = 128; s > 0; s >>= 1) {
    if (tid < s) { s1[tid] += s1[tid + s]; s2[tid] += s2[tid + s]; }
    __syncthreads();
  }
  if (tid == 0) {
    float n = (float)(M * Cc);
    float mu = s1[0] / n;
    float var = s2[0] / n - mu * mu;
    stats[b * 2 + 0] = mu;
    stats[b * 2 + 1] = fmaxf(var, 0.f);
  }
}

__global__ void gn_apply(const float* __restrict__ y,
                         const float* __restrict__ stats,
                         const float* __restrict__ g,
                         const float* __restrict__ bb,
                         const float* __restrict__ resid,
                         float* __restrict__ outp, int M) {
  int i = blockIdx.x * blockDim.x + threadIdx.x;
  if (i >= Bb * M * Cc) return;
  int b = i / (M * Cc), c = i % Cc;
  float mu = stats[b * 2], var = stats[b * 2 + 1];
  outp[i] = (y[i] - mu) * rsqrtf(var + EPSf) * g[c] + bb[c] + resid[i];
}

// =====================================================================
// Sparse BEV: pixel -> pillar-row map, then exact bilinear /2 /4 /8
// =====================================================================
__global__ void scatter_map(const int* __restrict__ coords,
                            int* __restrict__ map) {
  int t = blockIdx.x * blockDim.x + threadIdx.x;
  if (t >= Bb * Pp) return;
  const int* c = coords + (size_t)t * 4;
  int flat = c[1] + c[2] * NXx + c[3];
  map[(size_t)(t / Pp) * (NXx * NYy) + flat] = t;  // global pillar row
}

__global__ void resize_kernel(const int* __restrict__ map,
                              const float* __restrict__ ctxv,
                              float* __restrict__ outp, int H, int W, int s,
                              int o) {
  int i = blockIdx.x * blockDim.x + threadIdx.x;
  int total = Bb * Cc * H * W;
  if (i >= total) return;
  int ox = i % W;
  int t = i / W;
  int oy = t % H; t /= H;
  int c = t % Cc;
  int b = t / Cc;
  int y0 = s * oy + o, x0 = s * ox + o;
  const int* mp = map + (size_t)b * NXx * NYy;
  float a = 0.f;
#pragma unroll
  for (int dy = 0; dy < 2; ++dy)
#pragma unroll
    for (int dx = 0; dx < 2; ++dx) {
      int m = mp[(y0 + dy) * NXx + (x0 + dx)];
      if (m >= 0) a += ctxv[(size_t)m * Cc + c];
    }
  outp[i] = 0.25f * a;
}

// =====================================================================
// host orchestration
// =====================================================================
extern "C" void kernel_launch(void* const* d_in, const int* in_sizes, int n_in,
                              void* d_out, int out_size, void* d_ws,
                              size_t ws_size, hipStream_t stream) {
  (void)in_sizes; (void)n_in; (void)out_size; (void)ws_size;
  const float* pf     = (const float*)d_in[0];
  const int*   coords = (const int*)d_in[1];
  const float* w_off  = (const float*)d_in[2];
  const float* w1     = (const float*)d_in[3];
  const float* b1     = (const float*)d_in[4];
  const float* w2     = (const float*)d_in[5];
  const float* b2     = (const float*)d_in[6];
  const float* saw[3][6];
  for (int s = 0; s < 3; ++s)
    for (int j = 0; j < 6; ++j) saw[s][j] = (const float*)d_in[7 + s * 6 + j];

  char* base = (char*)d_ws;
  size_t off = 0;
  auto alloc = [&](size_t bytes) -> void* {
    off = (off + 255) & ~(size_t)255;
    void* p = base + off;
    off += bytes;
    return p;
  };

  float* centers = (float*)alloc(4ull * Bb * Pp * 3);
  float* kp      = (float*)alloc(4ull * Bb * Kk * 3);
  int*   kpidx   = (int*)  alloc(4ull * Bb * Kk);
  float* defxyz  = (float*)alloc(4ull * Bb * Kk * 3);
  int*   idx0    = (int*)  alloc(4ull * Bb * Kk * NSs);
  int*   val0    = (int*)  alloc(4ull * Bb * Kk * NSs);
  int*   idx1    = (int*)  alloc(4ull * Bb * Kk * NSs);
  int*   val1    = (int*)  alloc(4ull * Bb * Kk * NSs);
  float* local   = (float*)alloc(4ull * Bb * Kk * Cc);
  float* loc_sa  = (float*)alloc(4ull * Bb * Kk * Cc);
  float* a1      = (float*)alloc(4ull * Bb * Pp * Cc);
  float* ctx     = (float*)alloc(4ull * Bb * Pp * Cc);
  float* yf      = (float*)alloc(4ull * Bb * Pp * Cc);
  float* stats   = (float*)alloc(4ull * Bb * 2);
  int*   map     = (int*)  alloc(4ull * Bb * NXx * NYy);
  bf16*  xqb     = (bf16*) alloc(2ull * Bb * Pp * Cc);
  bf16*  xkb     = (bf16*) alloc(2ull * Bb * Kk * Cc);
  bf16*  qb      = (bf16*) alloc(2ull * Bb * Pp * Cc);
  bf16*  kb      = (bf16*) alloc(2ull * Bb * Kk * Cc);
  bf16*  vb      = (bf16*) alloc(2ull * Bb * Kk * Cc);
  bf16*  aob     = (bf16*) alloc(2ull * Bb * Pp * Cc);
  bf16*  wtb     = (bf16*) alloc(2ull * 4096);
  bf16*  wpb     = (bf16*) alloc(2ull * 4096);
  bf16*  wgb     = (bf16*) alloc(2ull * 4096);
  bf16*  wzb     = (bf16*) alloc(2ull * 4096);

  centers_kernel<<<(Bb * Pp + 255) / 256, 256, 0, stream>>>(coords, centers);
  fps_kernel<<<Bb, 1024, 0, stream>>>(centers, kpidx, kp);
  group_topk<<<dim3(Kk, 1, Bb), 256, 0, stream>>>(centers, kp, idx0, val0, DRAD2);
  deform_kernel<<<(Bb * Kk + 255) / 256, 256, 0, stream>>>(centers, pf, kp, idx0,
                                                           val0, w_off, defxyz);
  group_topk<<<dim3(Kk, 1, Bb), 256, 0, stream>>>(centers, defxyz, idx1, val1,
                                                  RAD2);
  mlp_kernel<<<dim3(Kk, 1, Bb), 64, 0, stream>>>(centers, pf, defxyz, idx1, val1,
                                                 w1, b1, w2, b2, local);

  auto conv = [&](const float* x, bf16* y, int nn) {
    f32_to_bf16<<<(nn + 255) / 256, 256, 0, stream>>>(x, y, nn);
  };
  auto gemm_bf = [&](const bf16* A, const bf16* W, bf16* Ob, int M) {
    int tiles = M >> 4;
    gemm64<false, true><<<dim3((tiles + 3) / 4, 1, Bb), 128, 0, stream>>>(
        A, W, nullptr, Ob, M);
  };
  auto gemm_f32 = [&](const bf16* A, const bf16* W, float* Of, int M) {
    int tiles = M >> 4;
    gemm64<true, false><<<dim3((tiles + 3) / 4, 1, Bb), 128, 0, stream>>>(
        A, W, Of, nullptr, M);
  };
  auto sa = [&](const float* qx, int Mq, const float* kv, int Mk,
                const float* const* ws2, float* outp) {
    conv(qx, xqb, Bb * Mq * Cc);
    conv(kv, xkb, Bb * Mk * Cc);
    conv(ws2[0], wtb, Cc * Cc);
    conv(ws2[1], wpb, Cc * Cc);
    conv(ws2[2], wgb, Cc * Cc);
    conv(ws2[3], wzb, Cc * Cc);
    gemm_bf(xqb, wtb, qb, Mq);
    gemm_bf(xkb, wpb, kb, Mk);
    gemm_bf(xkb, wgb, vb, Mk);
    int qt = Mq >> 4;
    flash_attn<<<dim3((qt + FAW - 1) / FAW, 1, Bb), 32 * FAW, 0, stream>>>(
        qb, kb, vb, aob, Mq, Mk);
    gemm_f32(aob, wzb, yf, Mq);
    gn_reduce<<<Bb, 256, 0, stream>>>(yf, stats, Mq);
    gn_apply<<<(Bb * Mq * Cc + 255) / 256, 256, 0, stream>>>(yf, stats, ws2[4],
                                                             ws2[5], qx, outp, Mq);
  };
  sa(local, Kk, local, Kk, saw[0], loc_sa);
  sa(pf, Pp, loc_sa, Kk, saw[1], a1);
  sa(a1, Pp, loc_sa, Kk, saw[2], ctx);

  (void)hipMemsetAsync(map, 0xFF, 4ull * Bb * NXx * NYy, stream);
  scatter_map<<<(Bb * Pp + 255) / 256, 256, 0, stream>>>(coords, map);

  float* out0 = (float*)d_out;
  float* out1 = out0 + (size_t)Bb * Cc * 248 * 216;
  float* out2 = out1 + (size_t)Bb * Cc * 124 * 108;
  resize_kernel<<<(Bb * Cc * 248 * 216 + 255) / 256, 256, 0, stream>>>(
      map, ctx, out0, 248, 216, 2, 0);
  resize_kernel<<<(Bb * Cc * 124 * 108 + 255) / 256, 256, 0, stream>>>(
      map, ctx, out1, 124, 108, 4, 1);
  resize_kernel<<<(Bb * Cc * 62 * 54 + 255) / 256, 256, 0, stream>>>(
      map, ctx, out2, 62, 54, 8, 3);
}
